// PointNet2Classification_53936199303449
// MI455X (gfx1250) — matrix operand order repaired and checked by reading, not compiled, not run on hardware
//
#include <hip/hip_runtime.h>
#include <cstdint>
#include <cstddef>

// ---------------------------------------------------------------------------
// PointNet++ MSG classification forward, CDNA5 (gfx1250, wave32, WMMA).
// GEMMs: v_wmma_f32_16x16x32_f16, one wave = 16x64 output strip (4 WMMA
// accumulators reusing one A tile). A/B tiles loaded as aligned b128 vectors
// per the ISA 7.12.2 f16 register layouts; weights pre-converted to padded
// f16 so the inner loop has zero predication. Activations staged f16
// f32 accumulation and
// f32 BN statistics.
// ---------------------------------------------------------------------------

typedef __attribute__((ext_vector_type(16))) _Float16 v16h;
typedef __attribute__((ext_vector_type(8)))  _Float16 v8h;
typedef __attribute__((ext_vector_type(8)))  float    v8f;

#define BATCH 8
#define NPTS  2048

// ------------------------------ utility kernels ----------------------------

__global__ void k_zero_f32(float* p, int n) {
  int i = blockIdx.x * blockDim.x + threadIdx.x;
  if (i < n) p[i] = 0.f;
}

// pointcloud [B,6,N] -> xyz [B,N,3], norm [B,N,3]
__global__ void k_split(const float* __restrict__ pc, float* __restrict__ xyz,
                        float* __restrict__ nrm, int Npts) {
  int e = blockIdx.x * blockDim.x + threadIdx.x;
  int tot = BATCH * Npts * 3;
  if (e >= tot) return;
  int b = e / (Npts * 3);
  int r = e - b * (Npts * 3);
  int n = r / 3, c = r - n * 3;
  xyz[e] = pc[(size_t)b * 6 * Npts + (size_t)c * Npts + n];
  nrm[e] = pc[(size_t)b * 6 * Npts + (size_t)(3 + c) * Npts + n];
}

// f32 [M,C] -> f16 [M,Cpad] zero-padded (also used for weight conversion)
__global__ void k_cvt_pad(const float* __restrict__ in, _Float16* __restrict__ out,
                          int M, int C, int Cpad) {
  size_t e = (size_t)blockIdx.x * blockDim.x + threadIdx.x;
  size_t tot = (size_t)M * Cpad;
  if (e >= tot) return;
  int c = (int)(e % Cpad);
  size_t row = e / Cpad;
  out[e] = (c < C) ? (_Float16)in[row * C + c] : (_Float16)0.f;
}

// ------------------------------ FPS ----------------------------------------
__global__ void k_fps(const float* __restrict__ xyz, int Npts, int npoint,
                      float* __restrict__ newxyz, int* __restrict__ outidx) {
  __shared__ float dist[2048];
  __shared__ float rv[256];
  __shared__ int   ri[256];
  int b = blockIdx.x, t = threadIdx.x;
  const float* base = xyz + (size_t)b * Npts * 3;
  for (int i = t; i < Npts; i += blockDim.x) dist[i] = 1e10f;
  __syncthreads();
  int far = 0;
  for (int s = 0; s < npoint; s++) {
    if (t == 0) {
      outidx[b * npoint + s] = far;
      newxyz[((size_t)b * npoint + s) * 3 + 0] = base[far * 3 + 0];
      newxyz[((size_t)b * npoint + s) * 3 + 1] = base[far * 3 + 1];
      newxyz[((size_t)b * npoint + s) * 3 + 2] = base[far * 3 + 2];
    }
    __syncthreads();
    float cx = base[far * 3 + 0], cy = base[far * 3 + 1], cz = base[far * 3 + 2];
    float bv = -1.f; int bi = 0;
    for (int i = t; i < Npts; i += blockDim.x) {
      float dx = base[i * 3 + 0] - cx;
      float dy = base[i * 3 + 1] - cy;
      float dz = base[i * 3 + 2] - cz;
      float d  = dx * dx + dy * dy + dz * dz;
      float nd = fminf(dist[i], d);
      dist[i] = nd;
      if (nd > bv || (nd == bv && i < bi)) { bv = nd; bi = i; }
    }
    rv[t] = bv; ri[t] = bi;
    __syncthreads();
    for (int o = 128; o > 0; o >>= 1) {
      if (t < o) {
        if (rv[t + o] > rv[t] || (rv[t + o] == rv[t] && ri[t + o] < ri[t])) {
          rv[t] = rv[t + o]; ri[t] = ri[t + o];
        }
      }
      __syncthreads();
    }
    far = ri[0];
  }
}

// ------------------------- ball query + grouping ---------------------------
__global__ void k_group(const float* __restrict__ xyz, const float* __restrict__ newxyz,
                        const float* __restrict__ feats, int Npts, int S, int K,
                        int D, int CinPad, float r2, _Float16* __restrict__ X) {
  __shared__ int idxs[128];
  __shared__ int mask[128];
  __shared__ int cnt;
  int g = blockIdx.x;
  int b = g / S;
  int t = threadIdx.x;
  const float* xb = xyz + (size_t)b * Npts * 3;
  float cx = newxyz[(size_t)g * 3 + 0];
  float cy = newxyz[(size_t)g * 3 + 1];
  float cz = newxyz[(size_t)g * 3 + 2];
  if (t == 0) cnt = 0;
  __syncthreads();
  for (int base = 0; base < Npts; base += 128) {
    int p = base + t;
    int m = 0;
    if (p < Npts) {
      float dx = xb[p * 3 + 0] - cx;
      float dy = xb[p * 3 + 1] - cy;
      float dz = xb[p * 3 + 2] - cz;
      m = (dx * dx + dy * dy + dz * dz <= r2) ? 1 : 0;
    }
    mask[t] = m;
    __syncthreads();
    if (t == 0) {
      int lim = (Npts - base < 128) ? (Npts - base) : 128;
      for (int i = 0; i < lim && cnt < K; i++)
        if (mask[i]) idxs[cnt++] = base + i;
    }
    __syncthreads();
    if (cnt >= K) break;
    __syncthreads();
  }
  if (t == 0) {
    int c = cnt;
    if (c == 0) { idxs[0] = 0; c = 1; }   // cannot happen (centroid in ball)
    for (int i = c; i < K; i++) idxs[i] = idxs[0];
  }
  __syncthreads();
  int tot = K * CinPad;
  for (int e = t; e < tot; e += 128) {
    int k = e / CinPad, c = e - k * CinPad;
    int pi = idxs[k];
    float v = 0.f;
    if (c < D) {
      v = feats[((size_t)b * Npts + pi) * D + c];
    } else if (c < D + 3) {
      int a = c - D;
      float cc = (a == 0) ? cx : ((a == 1) ? cy : cz);
      v = xb[pi * 3 + a] - cc;
    }
    X[(size_t)g * K * CinPad + e] = (_Float16)v;
  }
}

// group_all: rows=B*Npts2, cols = [xyz(3) | feats(D) | pad]
__global__ void k_groupall(const float* __restrict__ xyz2, const float* __restrict__ f2,
                           _Float16* __restrict__ X, int rows, int D, int Cpad) {
  size_t e = (size_t)blockIdx.x * blockDim.x + threadIdx.x;
  size_t tot = (size_t)rows * Cpad;
  if (e >= tot) return;
  int c = (int)(e % Cpad);
  size_t row = e / Cpad;
  float v = 0.f;
  if (c < 3)            v = xyz2[row * 3 + c];
  else if (c < 3 + D)   v = f2[row * D + (c - 3)];
  X[e] = (_Float16)v;
}

// ------------------------------- WMMA GEMM ---------------------------------
// Y[M,Cout](f16) = X[M,CinPad](f16 padded) @ Wh[Cout,CinPad](f16 padded)^T
// One wave computes a 16x64 strip: 1 A tile reused by 4 B tiles / 4 D accums.
// All tile loads are aligned 128-bit vector loads (zero predication).
__global__ void k_gemm_wmma(const _Float16* __restrict__ X, const _Float16* __restrict__ Wh,
                            const float* __restrict__ bias, _Float16* __restrict__ Y,
                            int M, int Cout, int CinPad, int mtiles) {
  int lane = threadIdx.x & 31;
  int wv   = threadIdx.x >> 5;
  int mt   = blockIdx.x * (blockDim.x >> 5) + wv;
  if (mt >= mtiles) return;
  int nt0  = blockIdx.y * 64;
  int hi   = (lane >> 4) & 1;

  int rowA = mt * 16 + (lane & 15);
  if (rowA >= M) rowA = M - 1;                 // clamp: only row rowA of D is affected
  const _Float16* xr = X + (size_t)rowA * CinPad;

  const _Float16* wr[4];
#pragma unroll
  for (int j = 0; j < 4; j++) {
    int cb = nt0 + j * 16 + (lane & 15);
    if (cb >= Cout) cb = Cout - 1;             // clamp: store is guarded
    wr[j] = Wh + (size_t)cb * CinPad;
  }

  v8f acc0 = {}, acc1 = {}, acc2 = {}, acc3 = {};

  for (int k0 = 0; k0 < CinPad; k0 += 32) {
    __builtin_prefetch(xr + k0 + 128, 0, 1);   // global_prefetch_b8

    // A tile: ISA f16 A-layout = two contiguous 8-half runs per lane
    v8h alo = *(const v8h*)(xr + k0 + hi * 8);
    v8h ahi = *(const v8h*)(xr + k0 + 16 + hi * 8);
    v16h a;
#pragma unroll
    for (int i = 0; i < 8; i++) { a[i] = alo[i]; a[8 + i] = ahi[i]; }

    // B tiles: same K order as A = two contiguous 8-half runs per lane
#pragma unroll
    for (int j = 0; j < 4; j++) {
      v8h blo = *(const v8h*)(wr[j] + k0 + hi * 8);
      v8h bhi = *(const v8h*)(wr[j] + k0 + 16 + hi * 8);
      v16h bm;
#pragma unroll
      for (int i = 0; i < 8; i++) { bm[i] = blo[i]; bm[8 + i] = bhi[i]; }
      v8f& acc = (j == 0) ? acc0 : (j == 1) ? acc1 : (j == 2) ? acc2 : acc3;
      acc = __builtin_amdgcn_wmma_f32_16x16x32_f16(
          false, a, false, bm, (short)0, acc, false, false);
    }
  }

#pragma unroll
  for (int j = 0; j < 4; j++) {
    int col = nt0 + j * 16 + (lane & 15);
    if (col >= Cout) continue;
    float bv = bias ? bias[col] : 0.f;
    v8f& acc = (j == 0) ? acc0 : (j == 1) ? acc1 : (j == 2) ? acc2 : acc3;
#pragma unroll
    for (int r = 0; r < 8; r++) {
      int row = mt * 16 + hi * 8 + r;          // ISA f32 C/D layout
      if (row < M) Y[(size_t)row * Cout + col] = (_Float16)(acc[r] + bv);
    }
  }
}

// -------------------- per-channel sum / sumsq reduction --------------------
__global__ void k_colsum(const _Float16* __restrict__ Y, int M, int C,
                         float* __restrict__ stats, int rowsPerBlock) {
  __shared__ float ls[1024];
  __shared__ float lq[1024];
  for (int c = threadIdx.x; c < C; c += blockDim.x) { ls[c] = 0.f; lq[c] = 0.f; }
  __syncthreads();
  int r0 = blockIdx.x * rowsPerBlock;
  int r1 = (r0 + rowsPerBlock < M) ? (r0 + rowsPerBlock) : M;
  const _Float16* base = Y + (size_t)r0 * C;
  size_t tot = (size_t)(r1 - r0) * C;
  for (size_t e = threadIdx.x; e < tot; e += blockDim.x) {
    float v = (float)base[e];
    int c = (int)(e % C);
    atomicAdd(&ls[c], v);        // ds_add_f32
    atomicAdd(&lq[c], v * v);
  }
  __syncthreads();
  for (int c = threadIdx.x; c < C; c += blockDim.x) {
    atomicAdd(&stats[c], ls[c]);           // global_atomic_add_f32
    atomicAdd(&stats[C + c], lq[c]);
  }
}

// -------- BN (batch stats) + ReLU, re-emit as next layer's padded f16 ------
__global__ void k_bnrelu(const _Float16* __restrict__ Y, int M, int C,
                         const float* __restrict__ stats,
                         const float* __restrict__ gamma, const float* __restrict__ beta,
                         _Float16* __restrict__ Xn, int CpadN, float invM) {
  size_t e = (size_t)blockIdx.x * blockDim.x + threadIdx.x;
  size_t tot = (size_t)M * CpadN;
  if (e >= tot) return;
  int c = (int)(e % CpadN);
  size_t row = e / CpadN;
  float o = 0.f;
  if (c < C) {
    float mean = stats[c] * invM;
    float var  = stats[C + c] * invM - mean * mean;
    float v    = (float)Y[row * C + c];
    o = gamma[c] * (v - mean) * rsqrtf(var + 1e-5f) + beta[c];
    o = fmaxf(o, 0.f);
  }
  Xn[e] = (_Float16)o;
}

// ------------------------------ max over K ---------------------------------
__global__ void k_maxpool(const _Float16* __restrict__ X, int K, int C, int Cpad,
                          float* __restrict__ dst, int Dtot, int coff) {
  int g = blockIdx.x;
  const _Float16* base = X + (size_t)g * K * Cpad;
  for (int c = threadIdx.x; c < C; c += blockDim.x) {
    float m = -1e30f;
    for (int k = 0; k < K; k++) m = fmaxf(m, (float)base[(size_t)k * Cpad + c]);
    dst[(size_t)g * Dtot + coff + c] = m;
  }
}

// ------------------------------ log_softmax --------------------------------
__global__ void k_logsoftmax(const _Float16* __restrict__ Y, float* __restrict__ out,
                             int M, int C) {
  int r = blockIdx.x * blockDim.x + threadIdx.x;
  if (r >= M) return;
  float mx = -1e30f;
  for (int c = 0; c < C; c++) mx = fmaxf(mx, (float)Y[r * C + c]);
  float s = 0.f;
  for (int c = 0; c < C; c++) s += expf((float)Y[r * C + c] - mx);
  float l = logf(s);
  for (int c = 0; c < C; c++) out[r * C + c] = (float)Y[r * C + c] - mx - l;
}

// ------------------------------ host driver --------------------------------

extern "C" void kernel_launch(void* const* d_in, const int* in_sizes, int n_in,
                              void* d_out, int out_size, void* d_ws, size_t ws_size,
                              hipStream_t stream) {
  (void)in_sizes; (void)n_in; (void)out_size; (void)ws_size;

  const float* pc = (const float*)d_in[0];
  int wi = 1;
  const float* sa1[3][3][3];
  for (int br = 0; br < 3; br++)
    for (int l = 0; l < 3; l++)
      for (int t = 0; t < 3; t++) sa1[br][l][t] = (const float*)d_in[wi++];
  const float* sa2[3][3][3];
  for (int br = 0; br < 3; br++)
    for (int l = 0; l < 3; l++)
      for (int t = 0; t < 3; t++) sa2[br][l][t] = (const float*)d_in[wi++];
  const float* sa3[3][3];
  for (int l = 0; l < 3; l++)
    for (int t = 0; t < 3; t++) sa3[l][t] = (const float*)d_in[wi++];
  const float* fc1W = (const float*)d_in[wi++];
  const float* fc1b = (const float*)d_in[wi++];
  const float* bn1g = (const float*)d_in[wi++];
  const float* bn1b = (const float*)d_in[wi++];
  const float* fc2W = (const float*)d_in[wi++];
  const float* fc2b = (const float*)d_in[wi++];
  const float* bn2g = (const float*)d_in[wi++];
  const float* bn2b = (const float*)d_in[wi++];
  const float* fc3W = (const float*)d_in[wi++];
  const float* fc3b = (const float*)d_in[wi++];

  // ---- workspace carve (bump allocator, 256B aligned) ----
  char* p = (char*)d_ws;
  auto alloc = [&](size_t bytes) -> char* {
    char* r = p;
    p += (bytes + 255) & ~(size_t)255;
    return r;
  };
  float* xyz   = (float*)alloc((size_t)BATCH * NPTS * 3 * 4);
  float* nrm   = (float*)alloc((size_t)BATCH * NPTS * 3 * 4);
  float* nx1   = (float*)alloc((size_t)BATCH * 512 * 3 * 4);
  int*   fpsi1 = (int*)  alloc((size_t)BATCH * 512 * 4);
  float* l1f   = (float*)alloc((size_t)BATCH * 512 * 320 * 4);
  float* nx2   = (float*)alloc((size_t)BATCH * 128 * 3 * 4);
  int*   fpsi2 = (int*)  alloc((size_t)BATCH * 128 * 4);
  float* l2f   = (float*)alloc((size_t)BATCH * 128 * 640 * 4);
  float* stats = (float*)alloc((size_t)2 * 1024 * 4);
  _Float16* Wh = (_Float16*)alloc((size_t)1024 * 1024 * 2);  // padded f16 weights
  const size_t XH_MAX = 67108864;  // 524288 rows x 128 cols (f16)
  const size_t YH_MAX = 67108864;
  _Float16* Xh = (_Float16*)alloc(XH_MAX * 2);
  _Float16* Yh = (_Float16*)alloc(YH_MAX * 2);

  auto pad32 = [](int x) { return (x + 31) & ~31; };

  auto run_gemm = [&](const _Float16* X, const float* W, const float* bias,
                      _Float16* Y, int M, int Cout, int Cin, int CinPad) {
    size_t wtot = (size_t)Cout * CinPad;
    k_cvt_pad<<<(unsigned)((wtot + 255) / 256), 256, 0, stream>>>(W, Wh, Cout, Cin,
                                                                  CinPad);
    int mtiles = (M + 15) / 16;
    dim3 grid((mtiles + 7) / 8, (Cout + 63) / 64);
    k_gemm_wmma<<<grid, 256, 0, stream>>>(X, Wh, bias, Y, M, Cout, CinPad, mtiles);
  };
  auto run_bn = [&](_Float16* Y, int M, int C, const float* g, const float* b,
                    _Float16* X, int CpadN) {
    k_zero_f32<<<(2 * C + 255) / 256, 256, 0, stream>>>(stats, 2 * C);
    int rpb = 512;
    k_colsum<<<(M + rpb - 1) / rpb, 256, 0, stream>>>(Y, M, C, stats, rpb);
    size_t tot = (size_t)M * CpadN;
    k_bnrelu<<<(unsigned)((tot + 255) / 256), 256, 0, stream>>>(
        Y, M, C, stats, g, b, X, CpadN, 1.f / (float)M);
  };

  // ---- split input ----
  {
    int tot = BATCH * NPTS * 3;
    k_split<<<(tot + 255) / 256, 256, 0, stream>>>(pc, xyz, nrm, NPTS);
  }

  // ---- SA1 (MSG): N=2048 -> S=512, feats = normals (D=3), out D=320 ----
  k_fps<<<BATCH, 256, 0, stream>>>(xyz, NPTS, 512, nx1, fpsi1);
  {
    const int mlps[3][3] = {{32, 32, 64}, {64, 64, 128}, {64, 96, 128}};
    const int Ks[3] = {16, 32, 128};
    const float rr[3] = {0.1f, 0.2f, 0.4f};
    int coff = 0;
    for (int br = 0; br < 3; br++) {
      int K = Ks[br], S = 512, M = BATCH * S * K, D = 3;
      int cin = D + 3, cpad = pad32(cin);
      k_group<<<BATCH * S, 128, 0, stream>>>(xyz, nx1, nrm, NPTS, S, K, D, cpad,
                                             rr[br] * rr[br], Xh);
      for (int l = 0; l < 3; l++) {
        int cout = mlps[br][l];
        run_gemm(Xh, sa1[br][l][0], nullptr, Yh, M, cout, cin, cpad);
        int cpN = (l < 2) ? pad32(cout) : cout;
        run_bn(Yh, M, cout, sa1[br][l][1], sa1[br][l][2], Xh, cpN);
        cin = cout; cpad = cpN;
      }
      k_maxpool<<<BATCH * S, 128, 0, stream>>>(Xh, K, cin, cpad, l1f, 320, coff);
      coff += cin;
    }
  }

  // ---- SA2 (MSG): N=512 -> S=128, feats = l1f (D=320), out D=640 ----
  k_fps<<<BATCH, 256, 0, stream>>>(nx1, 512, 128, nx2, fpsi2);
  {
    const int mlps[3][3] = {{64, 64, 128}, {128, 128, 256}, {128, 128, 256}};
    const int Ks[3] = {32, 64, 128};
    const float rr[3] = {0.2f, 0.4f, 0.8f};
    int coff = 0;
    for (int br = 0; br < 3; br++) {
      int K = Ks[br], S = 128, M = BATCH * S * K, D = 320;
      int cin = D + 3, cpad = pad32(cin);   // 323 -> 352
      k_group<<<BATCH * S, 128, 0, stream>>>(nx1, nx2, l1f, 512, S, K, D, cpad,
                                             rr[br] * rr[br], Xh);
      for (int l = 0; l < 3; l++) {
        int cout = mlps[br][l];
        run_gemm(Xh, sa2[br][l][0], nullptr, Yh, M, cout, cin, cpad);
        int cpN = (l < 2) ? pad32(cout) : cout;
        run_bn(Yh, M, cout, sa2[br][l][1], sa2[br][l][2], Xh, cpN);
        cin = cout; cpad = cpN;
      }
      k_maxpool<<<BATCH * S, 128, 0, stream>>>(Xh, K, cin, cpad, l2f, 640, coff);
      coff += cin;
    }
  }

  // ---- SA3 (group all): concat [xyz(3) | feats(640)] over 128 pts ----
  {
    int rows = BATCH * 128, cin = 643, cpad = 672;
    size_t tot = (size_t)rows * cpad;
    k_groupall<<<(unsigned)((tot + 255) / 256), 256, 0, stream>>>(nx2, l2f, Xh, rows,
                                                                 640, cpad);
    const int mlp[3] = {256, 512, 1024};
    int M = rows;
    for (int l = 0; l < 3; l++) {
      int cout = mlp[l];
      run_gemm(Xh, sa3[l][0], nullptr, Yh, M, cout, cin, cpad);
      int cpN = cout;  // already multiple of 32
      run_bn(Yh, M, cout, sa3[l][1], sa3[l][2], Xh, cpN);
      cin = cout; cpad = cpN;
    }
    // max over the 128 points -> l3 [B,1024], written straight into d_out[320:]
    float* l3 = (float*)d_out + 320;
    k_maxpool<<<BATCH, 128, 0, stream>>>(Xh, 128, 1024, 1024, l3, 1024, 0);
  }

  // ---- FC head: 1024 -> 512 -> 256 -> 40 (BN over batch, M=8) ----
  {
    const float* l3 = (const float*)d_out + 320;
    size_t tot = (size_t)8 * 1024;
    k_cvt_pad<<<(unsigned)((tot + 255) / 256), 256, 0, stream>>>(l3, Xh, 8, 1024, 1024);
    run_gemm(Xh, fc1W, fc1b, Yh, 8, 512, 1024, 1024);
    run_bn(Yh, 8, 512, bn1g, bn1b, Xh, 512);
    run_gemm(Xh, fc2W, fc2b, Yh, 8, 256, 512, 512);
    run_bn(Yh, 8, 256, bn2g, bn2b, Xh, 256);
    run_gemm(Xh, fc3W, fc3b, Yh, 8, 40, 256, 256);
    k_logsoftmax<<<1, 8, 0, stream>>>(Yh, (float*)d_out, 8, 40);
  }
}
